// NvWaveNet_90460601188467
// MI455X (gfx1250) — matrix operand order repaired and checked by reading, not compiled, not run on hardware
//
#include <hip/hip_runtime.h>

typedef __attribute__((ext_vector_type(16))) _Float16 v16h;
typedef __attribute__((ext_vector_type(8)))  _Float16 v8h;
typedef __attribute__((ext_vector_type(8)))  float    v8f;
typedef unsigned int u32x4 __attribute__((ext_vector_type(4)));
typedef int          i32x4 __attribute__((ext_vector_type(4)));
typedef int          i32x8 __attribute__((ext_vector_type(8)));

#define TSEQ   16384
#define BATCH  4
#define NLAYER 20
#define RES    64
#define GATE   128
#define SKIPC  256
#define OUTC   256
#define CIN    80
#define CINP   96
#define TT     64
#define SQRT_HALF 0.70710678118654752f

// ---------------------------------------------------------------------------
// WMMA fragment helpers (wave32, 16x16x32 f16 -> f32)
// A 16x32: lane = M (mod16); element j -> k = (j/8)*16 + (lane/16)*8 + j%8
// B 32x16: lane = N (mod16); element j -> k = (j/8)*16 + (lane/16)*8 + j%8
// C 16x16 f32: lane = N + 16*(M/8); vgpr v = M%8
// ---------------------------------------------------------------------------
__device__ __forceinline__ v16h load_frag_a(const _Float16* p, int ld, int lane) {
  const int row = lane & 15;
  const int kb  = (lane >> 4) << 3;              // 0 or 8
  const _Float16* q = p + row * ld + kb;
  v8h lo = *(const v8h*)(q);
  v8h hi = *(const v8h*)(q + 16);
  v16h r;
#pragma unroll
  for (int j = 0; j < 8; ++j) { r[j] = lo[j]; r[j + 8] = hi[j]; }
  return r;
}

__device__ __forceinline__ v16h load_frag_b(const _Float16* p, int ld, int lane) {
  const int col = lane & 15;
  const int kb  = (lane >> 4) << 3;              // 0 or 8
  const _Float16* q = p + col * ld + kb;
  v8h lo = *(const v8h*)(q);
  v8h hi = *(const v8h*)(q + 16);
  v16h r;
#pragma unroll
  for (int j = 0; j < 8; ++j) { r[j] = lo[j]; r[j + 8] = hi[j]; }
  return r;
}

__device__ __forceinline__ v8f wmma16(v16h a, v16h b, v8f c) {
  return __builtin_amdgcn_wmma_f32_16x16x32_f16(false, a, false, b, (short)0, c,
                                                false, false);
}

// ---------------------------------------------------------------------------
// Weight conversion: f32 -> f16 in WMMA-friendly layouts
// ---------------------------------------------------------------------------
__global__ void convert_weights(const float* __restrict__ Wd,  const float* __restrict__ Wc,
                                const float* __restrict__ Wsk, const float* __restrict__ Wo,
                                const float* __restrict__ Wl1, const float* __restrict__ Wl2,
                                _Float16* __restrict__ wd,  _Float16* __restrict__ wc,
                                _Float16* __restrict__ wsk, _Float16* __restrict__ wo,
                                _Float16* __restrict__ wl1, _Float16* __restrict__ wl2) {
  const int stride = gridDim.x * blockDim.x;
  const int i0 = blockIdx.x * blockDim.x + threadIdx.x;
  // W_dil [20][128][64][3] -> [20][3][128][64]
  for (int i = i0; i < NLAYER * 3 * GATE * RES; i += stride) {
    int r = i & 63, g = (i >> 6) & 127, k = (i >> 13) % 3, l = i / (3 << 13);
    wd[i] = (_Float16)Wd[(((size_t)l * GATE + g) * RES + r) * 3 + k];
  }
  // W_c [20][128][80] -> [20][128][96] zero-padded
  for (int i = i0; i < NLAYER * GATE * CINP; i += stride) {
    int ci = i % CINP, g = (i / CINP) & 127, l = i / (CINP * GATE);
    wc[i] = (ci < CIN) ? (_Float16)Wc[((size_t)l * GATE + g) * CIN + ci] : (_Float16)0.f;
  }
  for (int i = i0; i < NLAYER * SKIPC * RES; i += stride) wsk[i] = (_Float16)Wsk[i];
  for (int i = i0; i < NLAYER * RES * RES; i += stride)   wo[i]  = (_Float16)Wo[i];
  for (int i = i0; i < SKIPC * SKIPC; i += stride)        wl1[i] = (_Float16)Wl1[i];
  for (int i = i0; i < OUTC * SKIPC; i += stride)         wl2[i] = (_Float16)Wl2[i];
}

// c [4][80][T] f32 -> time-major padded f16 [4][T][96]
__global__ void convert_c(const float* __restrict__ c, _Float16* __restrict__ c16) {
  const int n = BATCH * TSEQ * CINP;
  for (int i = blockIdx.x * blockDim.x + threadIdx.x; i < n; i += gridDim.x * blockDim.x) {
    int ci = i % CINP;
    int bt = i / CINP;
    int t = bt & (TSEQ - 1);
    int b = bt >> 14;
    c16[i] = (ci < CIN) ? (_Float16)c[((size_t)b * CIN + ci) * TSEQ + t] : (_Float16)0.f;
  }
}

// ---------------------------------------------------------------------------
// First conv: kernel-2 causal, 256 -> 64 channels, tanh. VALU (3% of FLOPs).
// h32 stays [b][r][T]; h16 is time-major [b][T][64].
// ---------------------------------------------------------------------------
__global__ __launch_bounds__(256)
void first_conv(const float* __restrict__ x, const float* __restrict__ Wf,
                const float* __restrict__ bf, float* __restrict__ h32,
                _Float16* __restrict__ h16) {
  const int bb = blockIdx.y;
  const int t  = blockIdx.x * TT + (threadIdx.x & 63);
  const int r0 = (threadIdx.x >> 6) * 16;
  float acc[16];
#pragma unroll
  for (int j = 0; j < 16; ++j) acc[j] = bf[r0 + j];
  const bool has = (t > 0);
  for (int ch = 0; ch < OUTC; ++ch) {
    size_t xb = ((size_t)bb * OUTC + ch) * TSEQ + t;
    float x1 = x[xb];
    float x0 = has ? x[xb - 1] : 0.f;
#pragma unroll
    for (int j = 0; j < 16; ++j) {
      const float* w = Wf + ((size_t)(r0 + j) * OUTC + ch) * 2;
      acc[j] += w[0] * x0 + w[1] * x1;
    }
  }
#pragma unroll
  for (int j = 0; j < 16; ++j) {
    float hv = tanhf(acc[j]);
    h32[((size_t)bb * RES + r0 + j) * TSEQ + t] = hv;
    h16[((size_t)bb * TSEQ + t) * RES + r0 + j] = (_Float16)hv;
  }
}

// ---------------------------------------------------------------------------
// One WaveNet layer. h16/c16 are time-major, so LDS staging is pure 2D tile
// DMA: dilated taps via global_load_async_to_lds_b128 (ASYNCcnt), cond tile
// via the Tensor Data Mover (TENSORcnt). WMMA does all channel mixing.
// ---------------------------------------------------------------------------
__global__ __launch_bounds__(128)
void wavenet_layer(const float* __restrict__ h_in32, const _Float16* __restrict__ h_in16,
                   float* __restrict__ h_out32, _Float16* __restrict__ h_out16,
                   const _Float16* __restrict__ c16, float* __restrict__ skip,
                   const _Float16* __restrict__ Wd,  const float* __restrict__ bdil,
                   const _Float16* __restrict__ Wc,  const float* __restrict__ bcnd,
                   const _Float16* __restrict__ Wsk, const float* __restrict__ bsk,
                   const _Float16* __restrict__ Wo,  const float* __restrict__ bo,
                   int dilation, int layer) {
  // [tap][64 t][72] f16 + [64 t][96] f16, overlaid later with [64 t][132] f32
  __shared__ __align__(16) char smem[3 * TT * 72 * 2 + TT * CINP * 2];
  _Float16* sh_h = (_Float16*)smem;
  _Float16* sh_c = (_Float16*)(smem + 3 * TT * 72 * 2);
  float*    sh_y = (float*)smem;                    // reused after barrier
  __shared__ __align__(16) _Float16 sh_z[TT * 72];

  const int bb   = blockIdx.y;
  const int t0   = blockIdx.x * TT;
  const int tid  = threadIdx.x;
  const int wave = tid >> 5;
  const int lane = tid & 31;
  const int nloc = lane & 15;
  const int mh   = (lane >> 4) << 3;

  __builtin_prefetch(skip + ((size_t)bb * SKIPC) * TSEQ + t0, 0, 1);

  // ---- TDM: conditioning tile [64 t][96] as one tensor_load_to_lds ----
  if (tid == 0) {
    unsigned ldsa = (unsigned)(size_t)(void*)sh_c;
    unsigned long long ga =
        (unsigned long long)(size_t)(c16 + ((size_t)bb * TSEQ + t0) * CINP);
    u32x4 g0;
    g0[0] = 1u;                                    // count=1, user descriptor
    g0[1] = ldsa;                                  // lds_addr
    g0[2] = (unsigned)ga;                          // global_addr[31:0]
    g0[3] = (unsigned)(ga >> 32) | (2u << 30);     // global_addr[56:32] | type=2
    i32x8 g1;
    g1[0] = 1 << 16;                               // data_size = 2 bytes
    g1[1] = (CINP & 0xFFFF) << 16;                 // tensor_dim0[15:0] = 96
    g1[2] = 0;                                     // dim0 hi | dim1 lo (65536 -> 0)
    g1[3] = 1 | (CINP << 16);                      // dim1 hi = 1 | tile_dim0 = 96
    g1[4] = TT;                                    // tile_dim1 = 64, tile_dim2 = 0
    g1[5] = CINP;                                  // tensor_dim0_stride = 96
    g1[6] = 0;
    g1[7] = 0;
    i32x4 g2 = {0, 0, 0, 0};
    i32x4 g3 = {0, 0, 0, 0};
    asm volatile("tensor_load_to_lds %0, %1, %2, %3"
                 :: "s"(g0), "s"(g1), "s"(g2), "s"(g3) : "memory");
    __builtin_amdgcn_s_wait_tensorcnt(0);
  }

  // ---- async DMA: three dilated-tap tiles, rows are 128B in global ----
  const unsigned sh_h_base = (unsigned)(size_t)(void*)sh_h;
#pragma unroll
  for (int k = 0; k < 3; ++k) {
    const int dt = (2 - k) * dilation;
    for (int idx = tid; idx < TT * 8; idx += 128) {   // 8 x 16B chunks per row
      int tt = idx >> 3, ck = idx & 7;
      int t = t0 + tt - dt;
      if (t >= 0) {
        unsigned ldsa = sh_h_base + (unsigned)(k * TT * 144 + tt * 144 + ck * 16);
        const _Float16* g = h_in16 + ((size_t)bb * TSEQ + t) * RES + ck * 8;
        asm volatile("global_load_async_to_lds_b128 %0, %1, off"
                     :: "v"(ldsa), "v"(g) : "memory");
      } else {
        v8h z = {};
        *(v8h*)(sh_h + k * TT * 72 + tt * 72 + ck * 8) = z;   // causal zero pad
      }
    }
  }
  asm volatile("s_wait_asynccnt 0x0" ::: "memory");
  __syncthreads();

  // ---- gate GEMM: y[128 x 64t] = Wd(3 taps) @ h + Wc @ c + biases ----
  v8f acc[2][4];
#pragma unroll
  for (int mt = 0; mt < 2; ++mt) {
    int g0r = wave * 32 + mt * 16;
#pragma unroll
    for (int nt = 0; nt < 4; ++nt)
#pragma unroll
      for (int v = 0; v < 8; ++v)
        acc[mt][nt][v] = bdil[g0r + mh + v] + bcnd[g0r + mh + v];
  }
#pragma unroll
  for (int k = 0; k < 3; ++k) {
#pragma unroll
    for (int ks = 0; ks < 2; ++ks) {
      v16h a0 = load_frag_a(Wd + (size_t)(k * GATE + wave * 32) * RES + ks * 32, RES, lane);
      v16h a1 = load_frag_a(Wd + (size_t)(k * GATE + wave * 32 + 16) * RES + ks * 32, RES, lane);
#pragma unroll
      for (int nt = 0; nt < 4; ++nt) {
        v16h bf = load_frag_b(sh_h + k * TT * 72 + nt * 16 * 72 + ks * 32, 72, lane);
        acc[0][nt] = wmma16(a0, bf, acc[0][nt]);
        acc[1][nt] = wmma16(a1, bf, acc[1][nt]);
      }
    }
  }
#pragma unroll
  for (int ks = 0; ks < 3; ++ks) {
    v16h a0 = load_frag_a(Wc + (size_t)(wave * 32) * CINP + ks * 32, CINP, lane);
    v16h a1 = load_frag_a(Wc + (size_t)(wave * 32 + 16) * CINP + ks * 32, CINP, lane);
#pragma unroll
    for (int nt = 0; nt < 4; ++nt) {
      v16h bf = load_frag_b(sh_c + nt * 16 * CINP + ks * 32, CINP, lane);
      acc[0][nt] = wmma16(a0, bf, acc[0][nt]);
      acc[1][nt] = wmma16(a1, bf, acc[1][nt]);
    }
  }
  __syncthreads();   // input tiles dead; reuse smem as y

#pragma unroll
  for (int mt = 0; mt < 2; ++mt) {
    int g0r = wave * 32 + mt * 16;
#pragma unroll
    for (int nt = 0; nt < 4; ++nt)
#pragma unroll
      for (int v = 0; v < 8; ++v)
        sh_y[(nt * 16 + nloc) * 132 + g0r + mh + v] = acc[mt][nt][v];
  }
  __syncthreads();

  // ---- gated activation: z = tanh(a) * sigmoid(b) ----
  for (int idx = tid; idx < TT * RES; idx += 128) {
    int zc = idx >> 6, tt = idx & 63;
    float av = sh_y[tt * 132 + zc];
    float bv = sh_y[tt * 132 + RES + zc];
    float z  = tanhf(av) * (1.f / (1.f + __expf(-bv)));
    sh_z[tt * 72 + zc] = (_Float16)z;
  }
  __syncthreads();

  // ---- skip GEMM: s[256 x 64t] = Wsk @ z; accumulate into global skip ----
#pragma unroll
  for (int mt = 0; mt < 4; ++mt) {
    int s0 = wave * 64 + mt * 16;
    v8f sa[4];
#pragma unroll
    for (int nt = 0; nt < 4; ++nt)
#pragma unroll
      for (int v = 0; v < 8; ++v) sa[nt][v] = bsk[s0 + mh + v];
#pragma unroll
    for (int ks = 0; ks < 2; ++ks) {
      v16h a = load_frag_a(Wsk + (size_t)s0 * RES + ks * 32, RES, lane);
#pragma unroll
      for (int nt = 0; nt < 4; ++nt) {
        v16h bf = load_frag_b(sh_z + nt * 16 * 72 + ks * 32, 72, lane);
        sa[nt] = wmma16(a, bf, sa[nt]);
      }
    }
#pragma unroll
    for (int nt = 0; nt < 4; ++nt) {
      int t = t0 + nt * 16 + nloc;
#pragma unroll
      for (int v = 0; v < 8; ++v) {
        size_t ad = ((size_t)bb * SKIPC + s0 + mh + v) * TSEQ + t;
        float sv = sa[nt][v];
        skip[ad] = (layer == 0) ? sv : (skip[ad] + sv) * SQRT_HALF;
      }
    }
  }

  // ---- out GEMM: h' = (Wo @ z + bo + h) * sqrt(1/2) ----
  {
    int r0 = wave * 16;
    v8f oa[4];
#pragma unroll
    for (int nt = 0; nt < 4; ++nt)
#pragma unroll
      for (int v = 0; v < 8; ++v) oa[nt][v] = bo[r0 + mh + v];
#pragma unroll
    for (int ks = 0; ks < 2; ++ks) {
      v16h a = load_frag_a(Wo + (size_t)r0 * RES + ks * 32, RES, lane);
#pragma unroll
      for (int nt = 0; nt < 4; ++nt) {
        v16h bf = load_frag_b(sh_z + nt * 16 * 72 + ks * 32, 72, lane);
        oa[nt] = wmma16(a, bf, oa[nt]);
      }
    }
#pragma unroll
    for (int nt = 0; nt < 4; ++nt) {
      int t = t0 + nt * 16 + nloc;
#pragma unroll
      for (int v = 0; v < 8; ++v) {
        size_t ad = ((size_t)bb * RES + r0 + mh + v) * TSEQ + t;
        float hv = (oa[nt][v] + h_in32[ad]) * SQRT_HALF;
        h_out32[ad] = hv;
        h_out16[((size_t)bb * TSEQ + t) * RES + r0 + mh + v] = (_Float16)hv;
      }
    }
  }
}

// ---------------------------------------------------------------------------
// Final head: relu -> W_last1 -> relu -> W_last2, fused per 64-t tile.
// ---------------------------------------------------------------------------
__global__ __launch_bounds__(128)
void final_head(const float* __restrict__ skip,
                const _Float16* __restrict__ W1, const float* __restrict__ b1,
                const _Float16* __restrict__ W2, const float* __restrict__ b2,
                float* __restrict__ out) {
  __shared__ __align__(16) _Float16 sh[TT * 264];   // relu(skip), then u
  const int bb = blockIdx.y;
  const int t0 = blockIdx.x * TT;
  const int tid = threadIdx.x, wave = tid >> 5, lane = tid & 31;
  const int nloc = lane & 15, mh = (lane >> 4) << 3;

  for (int idx = tid; idx < SKIPC * TT; idx += 128) {
    int ch = idx >> 6, tt = idx & 63;
    float v = skip[((size_t)bb * SKIPC + ch) * TSEQ + t0 + tt];
    sh[tt * 264 + ch] = (_Float16)fmaxf(v, 0.f);
  }
  __syncthreads();

  v8f acc[4][4];
#pragma unroll
  for (int mt = 0; mt < 4; ++mt) {
    int o0 = wave * 64 + mt * 16;
#pragma unroll
    for (int nt = 0; nt < 4; ++nt)
#pragma unroll
      for (int v = 0; v < 8; ++v) acc[mt][nt][v] = b1[o0 + mh + v];
  }
#pragma unroll
  for (int mt = 0; mt < 4; ++mt) {
    int o0 = wave * 64 + mt * 16;
#pragma unroll
    for (int ks = 0; ks < 8; ++ks) {
      v16h a = load_frag_a(W1 + (size_t)o0 * SKIPC + ks * 32, SKIPC, lane);
#pragma unroll
      for (int nt = 0; nt < 4; ++nt) {
        v16h bf = load_frag_b(sh + nt * 16 * 264 + ks * 32, 264, lane);
        acc[mt][nt] = wmma16(a, bf, acc[mt][nt]);
      }
    }
  }
  __syncthreads();
#pragma unroll
  for (int mt = 0; mt < 4; ++mt) {
    int o0 = wave * 64 + mt * 16;
#pragma unroll
    for (int nt = 0; nt < 4; ++nt)
#pragma unroll
      for (int v = 0; v < 8; ++v)
        sh[(nt * 16 + nloc) * 264 + o0 + mh + v] = (_Float16)fmaxf(acc[mt][nt][v], 0.f);
  }
  __syncthreads();

#pragma unroll
  for (int mt = 0; mt < 4; ++mt) {
    int o0 = wave * 64 + mt * 16;
#pragma unroll
    for (int nt = 0; nt < 4; ++nt)
#pragma unroll
      for (int v = 0; v < 8; ++v) acc[mt][nt][v] = b2[o0 + mh + v];
  }
#pragma unroll
  for (int mt = 0; mt < 4; ++mt) {
    int o0 = wave * 64 + mt * 16;
#pragma unroll
    for (int ks = 0; ks < 8; ++ks) {
      v16h a = load_frag_a(W2 + (size_t)o0 * SKIPC + ks * 32, SKIPC, lane);
#pragma unroll
      for (int nt = 0; nt < 4; ++nt) {
        v16h bf = load_frag_b(sh + nt * 16 * 264 + ks * 32, 264, lane);
        acc[mt][nt] = wmma16(a, bf, acc[mt][nt]);
      }
    }
  }
#pragma unroll
  for (int mt = 0; mt < 4; ++mt) {
    int o0 = wave * 64 + mt * 16;
#pragma unroll
    for (int nt = 0; nt < 4; ++nt) {
      int t = t0 + nt * 16 + nloc;
#pragma unroll
      for (int v = 0; v < 8; ++v)
        out[((size_t)bb * OUTC + o0 + mh + v) * TSEQ + t] = acc[mt][nt][v];
    }
  }
}

// ---------------------------------------------------------------------------
extern "C" void kernel_launch(void* const* d_in, const int* in_sizes, int n_in,
                              void* d_out, int out_size, void* d_ws, size_t ws_size,
                              hipStream_t stream) {
  (void)in_sizes; (void)n_in; (void)out_size; (void)ws_size;
  const float* x     = (const float*)d_in[0];
  const float* c     = (const float*)d_in[1];
  const float* Wf    = (const float*)d_in[2];
  const float* bf    = (const float*)d_in[3];
  const float* Wdil  = (const float*)d_in[4];
  const float* bdil  = (const float*)d_in[5];
  const float* Wc    = (const float*)d_in[6];
  const float* bc    = (const float*)d_in[7];
  const float* Wskip = (const float*)d_in[8];
  const float* bskip = (const float*)d_in[9];
  const float* Wout  = (const float*)d_in[10];
  const float* bout  = (const float*)d_in[11];
  const float* Wl1   = (const float*)d_in[12];
  const float* bl1   = (const float*)d_in[13];
  const float* Wl2   = (const float*)d_in[14];
  const float* bl2   = (const float*)d_in[15];

  char* ws = (char*)d_ws;
  size_t off = 0;
  auto take = [&](size_t bytes) {
    void* p = ws + off;
    off = (off + bytes + 255) & ~(size_t)255;
    return p;
  };
  float*    h32[2]; _Float16* h16[2];
  h32[0] = (float*)take((size_t)BATCH * RES * TSEQ * 4);
  h32[1] = (float*)take((size_t)BATCH * RES * TSEQ * 4);
  h16[0] = (_Float16*)take((size_t)BATCH * TSEQ * RES * 2);
  h16[1] = (_Float16*)take((size_t)BATCH * TSEQ * RES * 2);
  float*    skipb = (float*)take((size_t)BATCH * SKIPC * TSEQ * 4);
  _Float16* c16   = (_Float16*)take((size_t)BATCH * TSEQ * CINP * 2);
  _Float16* wd    = (_Float16*)take((size_t)NLAYER * 3 * GATE * RES * 2);
  _Float16* wc16  = (_Float16*)take((size_t)NLAYER * GATE * CINP * 2);
  _Float16* wsk   = (_Float16*)take((size_t)NLAYER * SKIPC * RES * 2);
  _Float16* wo    = (_Float16*)take((size_t)NLAYER * RES * RES * 2);
  _Float16* wl1   = (_Float16*)take((size_t)SKIPC * SKIPC * 2);
  _Float16* wl2   = (_Float16*)take((size_t)OUTC * SKIPC * 2);

  convert_weights<<<512, 256, 0, stream>>>(Wdil, Wc, Wskip, Wout, Wl1, Wl2,
                                           wd, wc16, wsk, wo, wl1, wl2);
  convert_c<<<4096, 256, 0, stream>>>(c, c16);

  dim3 grid(TSEQ / TT, BATCH);
  first_conv<<<grid, 256, 0, stream>>>(x, Wf, bf, h32[0], h16[0]);

  for (int l = 0; l < NLAYER; ++l) {
    int d = 1 << (l % (NLAYER / 2));
    wavenet_layer<<<grid, 128, 0, stream>>>(
        h32[l & 1], h16[l & 1], h32[(l + 1) & 1], h16[(l + 1) & 1],
        c16, skipb,
        wd + (size_t)l * 3 * GATE * RES, bdil + (size_t)l * GATE,
        wc16 + (size_t)l * GATE * CINP, bc + (size_t)l * GATE,
        wsk + (size_t)l * SKIPC * RES, bskip + (size_t)l * SKIPC,
        wo + (size_t)l * RES * RES, bout + (size_t)l * RES,
        d, l);
  }
  final_head<<<grid, 128, 0, stream>>>(skipb, wl1, bl1, wl2, bl2, (float*)d_out);
}
